// MSAColumnAttention_20856361189898
// MI455X (gfx1250) — matrix operand order repaired and checked by reading, not compiled, not run on hardware
//
#include <hip/hip_runtime.h>
#include <hip/hip_bf16.h>
#include <math.h>

// ---------------------------------------------------------------------------
// MSA column attention for MI455X (gfx1250), wave32 + WMMA bf16 path.
// B=1, S(seq)=256, L(res)=192, D=256, H=8, C=32.
// ---------------------------------------------------------------------------

#define NSEQ 256
#define NRES 192
#define DMODEL 256
#define NHEAD 8
#define CHEAD 32
#define NEG_INF_F (-10000.0f)
#define LN_EPS_F 1e-5f
#define QSCALE 0.17677669529663689f /* 1/sqrt(32) */

typedef __attribute__((ext_vector_type(16))) __bf16 bf16x16;
typedef __attribute__((ext_vector_type(8))) float f32x8;

// ---------------------------------------------------------------------------
// Async global -> LDS copy (CDNA5 GLOBAL_LOAD_ASYNC_TO_LDS_B128, ASYNCcnt).
// Builtin exists on this toolchain; parameter 1 is an int4-vector pointer in
// the global address space (clang prints it as `v4i __device__*`), so we cast
// through integers to the exact addrspace-qualified vector pointer types.
// ---------------------------------------------------------------------------
#if defined(__HIP_DEVICE_COMPILE__) && \
    __has_builtin(__builtin_amdgcn_global_load_async_to_lds_b128)
#define USE_ASYNC_LDS 1
#else
#define USE_ASYNC_LDS 0
#endif

typedef int v4i __attribute__((vector_size(16)));
typedef __attribute__((address_space(1))) v4i as1_v4i;
typedef __attribute__((address_space(3))) v4i as3_v4i;

__device__ __forceinline__ void copy16_g2l(void* lds_dst, const void* gsrc) {
#if USE_ASYNC_LDS
  // generic LDS pointer low 32 bits == LDS byte offset (aperture model)
  __builtin_amdgcn_global_load_async_to_lds_b128(
      (as1_v4i*)(uintptr_t)gsrc,
      (as3_v4i*)(unsigned int)(uintptr_t)lds_dst, 0, 0);
#else
  *(uint4*)lds_dst = *(const uint4*)gsrc;
#endif
}

__device__ __forceinline__ void wait_async_copies() {
#if USE_ASYNC_LDS
#if __has_builtin(__builtin_amdgcn_s_wait_asynccnt)
  __builtin_amdgcn_s_wait_asynccnt(0);
#endif
#endif
}

__device__ __forceinline__ f32x8 zero8() {
  f32x8 z;
#pragma unroll
  for (int i = 0; i < 8; ++i) z[i] = 0.0f;
  return z;
}

// A fragment: 16x32 bf16 (MxK), row-major source with leading dim ld.
// Lane layout (ISA 7.12.2): lane = hi*16 + m; elements 0..7 -> K = 8*hi+e,
// elements 8..15 -> K = 16 + 8*hi + (e-8).
__device__ __forceinline__ bf16x16 load_a_frag(const __bf16* base, int ld) {
  const int lane = threadIdx.x & 31;
  const int m = lane & 15;
  const int hi = lane >> 4;
  const __bf16* p = base + (size_t)m * ld;
  bf16x16 a;
#pragma unroll
  for (int e = 0; e < 8; ++e) a[e] = p[8 * hi + e];
#pragma unroll
  for (int e = 0; e < 8; ++e) a[8 + e] = p[16 + 8 * hi + e];
  return a;
}

// B fragment: 32x16 bf16 (KxN), row-major source with leading dim ld.
// Lane layout: lane = hi*16 + n; elements 0..7 -> K = 8*hi+e,
// elements 8..15 -> K = 16 + 8*hi + (e-8).
__device__ __forceinline__ bf16x16 load_b_frag(const __bf16* base, int ld) {
  const int lane = threadIdx.x & 31;
  const int n = lane & 15;
  const int hi = lane >> 4;
  const __bf16* p = base + n;
  bf16x16 b;
#pragma unroll
  for (int e = 0; e < 8; ++e) b[e] = p[(size_t)(8 * hi + e) * ld];
#pragma unroll
  for (int e = 0; e < 8; ++e) b[8 + e] = p[(size_t)(16 + 8 * hi + e) * ld];
  return b;
}

__device__ __forceinline__ f32x8 wmma_bf16(bf16x16 a, bf16x16 b, f32x8 c) {
  return __builtin_amdgcn_wmma_f32_16x16x32_bf16(
      /*neg_a=*/false, a, /*neg_b=*/false, b,
      /*c_mod=*/(short)0, c, /*reuse_a=*/false, /*reuse_b=*/false);
}

// ---------------------------------------------------------------------------
// Kernel 0: one-time weight prep. Converts the five DxD f32 weights to
// transposed bf16 planes wT[plane][d][j] = W[j][d] in workspace.
// grid = 5*256 (plane,d), 256 threads (j).
// ---------------------------------------------------------------------------
__global__ void msa_prep_weights_kernel(const float* __restrict__ wq,
                                        const float* __restrict__ wk,
                                        const float* __restrict__ wv,
                                        const float* __restrict__ wg,
                                        const float* __restrict__ wo,
                                        __bf16* __restrict__ wTall) {
  const int plane = blockIdx.x >> 8;
  const int d = blockIdx.x & 255;
  const int j = threadIdx.x;
  const float* Ws[5] = {wq, wk, wv, wg, wo};
  wTall[((size_t)plane * DMODEL + d) * DMODEL + j] =
      (__bf16)Ws[plane][(size_t)j * DMODEL + d];
}

// ---------------------------------------------------------------------------
// Kernel 1: LayerNorm + Q/K/V/G projections.
// grid = NRES*4 blocks, 128 threads (4 waves), each block: one l, 64 s-rows.
// dyn LDS: xn[64][256] bf16 (32KB) + wtile[256][32] bf16 (16KB) = 48KB.
// Weight tiles arrive via async global->LDS from pre-converted bf16 planes.
// ---------------------------------------------------------------------------
__global__ void msa_ln_qkvg_kernel(const float* __restrict__ m,
                                   const float* __restrict__ ln_scale,
                                   const float* __restrict__ ln_bias,
                                   const __bf16* __restrict__ wTall,
                                   const float* __restrict__ bg,
                                   __bf16* __restrict__ qws,
                                   __bf16* __restrict__ kws,
                                   __bf16* __restrict__ vws,
                                   __bf16* __restrict__ gws) {
  extern __shared__ char smem[];
  __bf16* xn = (__bf16*)smem;                      // [64][256]
  __bf16* wtile = (__bf16*)(smem + 64 * 256 * 2);  // [256][32] (d-major)

  const int l = blockIdx.x >> 2;
  const int s0 = (blockIdx.x & 3) * 64;
  const int tid = threadIdx.x;
  const int w = tid >> 5;
  const int lane = tid & 31;

  // ---- Phase A: LayerNorm 16 rows per wave, write bf16 xn to LDS ----
#pragma unroll 1
  for (int rr = 0; rr < 16; ++rr) {
    const int s = s0 + w * 16 + rr;
    const float* row = m + ((size_t)s * NRES + l) * DMODEL;
    float x[8];
    float sum = 0.0f, sq = 0.0f;
#pragma unroll
    for (int i = 0; i < 8; ++i) {
      x[i] = row[lane * 8 + i];
      sum += x[i];
      sq += x[i] * x[i];
    }
#pragma unroll
    for (int msk = 1; msk < 32; msk <<= 1) {
      sum += __shfl_xor(sum, msk, 32);
      sq += __shfl_xor(sq, msk, 32);
    }
    const float mu = sum * (1.0f / DMODEL);
    const float var = sq * (1.0f / DMODEL) - mu * mu;
    const float rstd = rsqrtf(var + LN_EPS_F);
#pragma unroll
    for (int i = 0; i < 8; ++i) {
      const int d = lane * 8 + i;
      const float v = (x[i] - mu) * rstd * ln_scale[d] + ln_bias[d];
      xn[(w * 16 + rr) * DMODEL + d] = (__bf16)v;
    }
  }
  __syncthreads();

  // ---- Phase B: four projections, 32-output-column chunks ----
  __bf16* Os[4] = {qws, kws, vws, gws};

#pragma unroll 1
  for (int wi = 0; wi < 4; ++wi) {
    const __bf16* wplane = wTall + (size_t)wi * DMODEL * DMODEL;  // [d][j]
    __bf16* O = Os[wi];
#pragma unroll 1
    for (int jc = 0; jc < 8; ++jc) {
      // stage wtile[d][jj] = wplane[d][jc*32+jj]; 1024 x 16B async moves
      for (int idx = tid; idx < DMODEL * 4; idx += 128) {
        const int d = idx >> 2;
        const int part = idx & 3;  // 8 bf16 = 16 bytes per part
        copy16_g2l(&wtile[d * 32 + part * 8],
                   &wplane[(size_t)d * DMODEL + jc * 32 + part * 8]);
      }
      wait_async_copies();
      __syncthreads();

      f32x8 acc0 = zero8(), acc1 = zero8();
#pragma unroll
      for (int kc = 0; kc < 8; ++kc) {
        bf16x16 a = load_a_frag(&xn[(w * 16) * DMODEL + kc * 32], DMODEL);
        bf16x16 b0 = load_b_frag(&wtile[(kc * 32) * 32 + 0], 32);
        acc0 = wmma_bf16(a, b0, acc0);
        bf16x16 b1 = load_b_frag(&wtile[(kc * 32) * 32 + 16], 32);
        acc1 = wmma_bf16(a, b1, acc1);
      }
      __syncthreads();  // wtile is restaged next iteration

      // epilogue: h == jc because column chunk width == CHEAD
      const int n = lane & 15;
      const int hi = lane >> 4;
#pragma unroll
      for (int nc = 0; nc < 2; ++nc) {
        f32x8 acc = nc ? acc1 : acc0;
#pragma unroll
        for (int r = 0; r < 8; ++r) {
          const int s = s0 + w * 16 + r + 8 * hi;
          const int c = nc * 16 + n;
          const int d = jc * 32 + c;
          float val = acc[r];
          if (wi == 0) val *= QSCALE;
          if (wi == 3) val = 1.0f / (1.0f + __expf(-(val + bg[d])));
          O[(((size_t)(l * NHEAD + jc)) * NSEQ + s) * CHEAD + c] = (__bf16)val;
        }
      }
    }
  }
}

// ---------------------------------------------------------------------------
// Kernel 2: attention per (l, h). grid = NRES*NHEAD, 256 threads (8 waves).
// dyn LDS: kT[32][256] bf16 (16KB) + v[256][32] bf16 (16KB)
//        + pbuf[8][16][256] bf16 (64KB) + bias[256] f32 (1KB) = 97KB
// (legal: CDNA5 allows up to 320KB LDS per workgroup).
// ---------------------------------------------------------------------------
__global__ void msa_attn_kernel(const __bf16* __restrict__ qws,
                                const __bf16* __restrict__ kws,
                                const __bf16* __restrict__ vws,
                                const __bf16* __restrict__ gws,
                                const unsigned char* __restrict__ seq_pad,
                                const unsigned char* __restrict__ res_pad,
                                __bf16* __restrict__ ctx) {
  extern __shared__ char smem[];
  __bf16* kT = (__bf16*)smem;                               // [32][256]
  __bf16* vbuf = (__bf16*)(smem + 32 * 256 * 2);            // [256][32]
  __bf16* pbuf = (__bf16*)(smem + 2 * 32 * 256 * 2);        // [8][16][256]
  float* biasv = (float*)(smem + 2 * 32 * 256 * 2 + 8 * 16 * 256 * 2);  // [256]

  const int l = blockIdx.x >> 3;
  const int h = blockIdx.x & 7;
  const int tid = threadIdx.x;
  const int w = tid >> 5;
  const int lane = tid & 31;
  const size_t lh = (size_t)(l * NHEAD + h) * NSEQ * CHEAD;

  // V tile: identical layout in LDS and workspace -> straight async copy
  for (int idx = tid; idx < NSEQ * CHEAD / 8; idx += 256)  // 1024 x 16B
    copy16_g2l(&vbuf[idx * 8], &vws[lh + (size_t)idx * 8]);

  // K^T tile needs a transpose -> manual staging (overlaps with async V copy)
  for (int idx = tid; idx < NSEQ * CHEAD; idx += 256) {
    const int t = idx >> 5;
    const int c = idx & 31;
    kT[c * NSEQ + t] = kws[lh + idx];
  }
  if (tid < NSEQ)
    biasv[tid] = (res_pad[l] || seq_pad[tid]) ? NEG_INF_F : 0.0f;
  wait_async_copies();
  __syncthreads();

  const int n = lane & 15;
  const int hi = lane >> 4;
  __bf16* pb = pbuf + (size_t)w * 16 * NSEQ;

#pragma unroll 1
  for (int si = 0; si < 2; ++si) {
    const int s0 = (w + si * 8) * 16;  // query slab base

    // Q fragment: 16 query rows x 32 channels (full K-dim in one fragment)
    bf16x16 aq = load_a_frag(qws + lh + (size_t)s0 * CHEAD, CHEAD);

    // ---- logits = q @ k^T + bias ----
    f32x8 acc[16];
#pragma unroll
    for (int tt = 0; tt < 16; ++tt) {
      bf16x16 bk = load_b_frag(kT + tt * 16, NSEQ);
      acc[tt] = wmma_bf16(aq, bk, zero8());
      const float bl = biasv[tt * 16 + n];
#pragma unroll
      for (int r = 0; r < 8; ++r) acc[tt][r] += bl;
    }

    // ---- softmax over 256 keys, in registers ----
    // C/D layout: element r of lane holds row (r + 8*hi), col (16*tt + n);
    // a full logits row lives in lanes {hi*16 .. hi*16+15} -> 4-step xor reduce.
    f32x8 rmax = acc[0];
#pragma unroll
    for (int tt = 1; tt < 16; ++tt)
#pragma unroll
      for (int r = 0; r < 8; ++r) rmax[r] = fmaxf(rmax[r], acc[tt][r]);
#pragma unroll
    for (int r = 0; r < 8; ++r) {
#pragma unroll
      for (int msk = 1; msk < 16; msk <<= 1)
        rmax[r] = fmaxf(rmax[r], __shfl_xor(rmax[r], msk, 32));
    }
    f32x8 rsum = zero8();
#pragma unroll
    for (int tt = 0; tt < 16; ++tt)
#pragma unroll
      for (int r = 0; r < 8; ++r) {
        acc[tt][r] = __expf(acc[tt][r] - rmax[r]);
        rsum[r] += acc[tt][r];
      }
#pragma unroll
    for (int r = 0; r < 8; ++r) {
#pragma unroll
      for (int msk = 1; msk < 16; msk <<= 1)
        rsum[r] += __shfl_xor(rsum[r], msk, 32);
      rsum[r] = 1.0f / rsum[r];
    }

    // probs -> LDS as bf16 (per-wave buffer; same-wave DS ops are in-order)
#pragma unroll
    for (int tt = 0; tt < 16; ++tt)
#pragma unroll
      for (int r = 0; r < 8; ++r)
        pb[(r + 8 * hi) * NSEQ + tt * 16 + n] = (__bf16)(acc[tt][r] * rsum[r]);

    // ---- ctx = probs @ v, gated, store bf16 ----
#pragma unroll
    for (int nc = 0; nc < 2; ++nc) {
      f32x8 o = zero8();
#pragma unroll
      for (int kc = 0; kc < 8; ++kc) {
        bf16x16 ap = load_a_frag(pb + kc * 32, NSEQ);
        bf16x16 bv = load_b_frag(vbuf + (kc * 32) * CHEAD + nc * 16, CHEAD);
        o = wmma_bf16(ap, bv, o);
      }
#pragma unroll
      for (int r = 0; r < 8; ++r) {
        const int s = s0 + r + 8 * hi;
        const int c = nc * 16 + n;
        const float gv = (float)gws[lh + (size_t)s * CHEAD + c];
        ctx[((size_t)l * NSEQ + s) * DMODEL + h * CHEAD + c] =
            (__bf16)(o[r] * gv);
      }
    }
  }
}

// ---------------------------------------------------------------------------
// Kernel 3: out = ctx @ wo^T + bo, write f32 in (S, L, D) order.
// grid = NRES*4, 128 threads (4 waves). dyn LDS: wtile[256][32] bf16 (16KB).
// ---------------------------------------------------------------------------
__global__ void msa_outproj_kernel(const __bf16* __restrict__ ctx,
                                   const __bf16* __restrict__ woT,
                                   const float* __restrict__ bo,
                                   float* __restrict__ out) {
  extern __shared__ char smem[];
  __bf16* wtile = (__bf16*)smem;  // [256][32]

  const int l = blockIdx.x >> 2;
  const int s0 = (blockIdx.x & 3) * 64;
  const int tid = threadIdx.x;
  const int w = tid >> 5;
  const int lane = tid & 31;
  const int n = lane & 15;
  const int hi = lane >> 4;

#pragma unroll 1
  for (int jc = 0; jc < 8; ++jc) {
    for (int idx = tid; idx < DMODEL * 4; idx += 128) {
      const int d = idx >> 2;
      const int part = idx & 3;
      copy16_g2l(&wtile[d * 32 + part * 8],
                 &woT[(size_t)d * DMODEL + jc * 32 + part * 8]);
    }
    wait_async_copies();
    __syncthreads();

    f32x8 acc0 = zero8(), acc1 = zero8();
#pragma unroll
    for (int kc = 0; kc < 8; ++kc) {
      bf16x16 a = load_a_frag(
          ctx + ((size_t)l * NSEQ + s0 + w * 16) * DMODEL + kc * 32, DMODEL);
      bf16x16 b0 = load_b_frag(&wtile[(kc * 32) * 32 + 0], 32);
      acc0 = wmma_bf16(a, b0, acc0);
      bf16x16 b1 = load_b_frag(&wtile[(kc * 32) * 32 + 16], 32);
      acc1 = wmma_bf16(a, b1, acc1);
    }
    __syncthreads();

#pragma unroll
    for (int nc = 0; nc < 2; ++nc) {
      f32x8 acc = nc ? acc1 : acc0;
#pragma unroll
      for (int r = 0; r < 8; ++r) {
        const int s = s0 + w * 16 + r + 8 * hi;
        const int d = jc * 32 + nc * 16 + n;
        out[((size_t)s * NRES + l) * DMODEL + d] = acc[r] + bo[d];
      }
    }
  }
}

// ---------------------------------------------------------------------------
// Host launcher
// ---------------------------------------------------------------------------
extern "C" void kernel_launch(void* const* d_in, const int* in_sizes, int n_in,
                              void* d_out, int out_size, void* d_ws,
                              size_t ws_size, hipStream_t stream) {
  const float* m = (const float*)d_in[0];
  const unsigned char* seq_pad = (const unsigned char*)d_in[1];
  const unsigned char* res_pad = (const unsigned char*)d_in[2];
  const float* ln_scale = (const float*)d_in[3];
  const float* ln_bias = (const float*)d_in[4];
  const float* wq = (const float*)d_in[5];
  const float* wk = (const float*)d_in[6];
  const float* wv = (const float*)d_in[7];
  const float* wg = (const float*)d_in[8];
  const float* bg = (const float*)d_in[9];
  const float* wo = (const float*)d_in[10];
  const float* bo = (const float*)d_in[11];
  float* out = (float*)d_out;

  // workspace layout:
  //   [0, 1MB): 5 transposed bf16 weight planes (5 * 256*256 * 2 = 640KB)
  //   then q, k, v, g, ctx — bf16, 49152x256 each
  char* ws = (char*)d_ws;
  __bf16* wTall = (__bf16*)ws;
  const size_t WOFF = (size_t)1 << 20;
  const size_t P = (size_t)NRES * NSEQ * DMODEL * sizeof(__bf16);
  __bf16* qws = (__bf16*)(ws + WOFF + 0 * P);
  __bf16* kws = (__bf16*)(ws + WOFF + 1 * P);
  __bf16* vws = (__bf16*)(ws + WOFF + 2 * P);
  __bf16* gws = (__bf16*)(ws + WOFF + 3 * P);
  __bf16* ctx = (__bf16*)(ws + WOFF + 4 * P);

  const size_t smem1 = (size_t)64 * 256 * 2 + (size_t)256 * 32 * 2;  // 48KB
  const size_t smem2 = (size_t)2 * 32 * 256 * 2 + (size_t)8 * 16 * 256 * 2 +
                       (size_t)256 * 4;  // 97KB (CDNA5 LDS allows 320KB/WG)
  const size_t smem3 = (size_t)256 * 32 * 2;  // 16KB

  msa_prep_weights_kernel<<<dim3(5 * 256), dim3(256), 0, stream>>>(
      wq, wk, wv, wg, wo, wTall);

  msa_ln_qkvg_kernel<<<dim3(NRES * 4), dim3(128), smem1, stream>>>(
      m, ln_scale, ln_bias, wTall, bg, qws, kws, vws, gws);

  msa_attn_kernel<<<dim3(NRES * NHEAD), dim3(256), smem2, stream>>>(
      qws, kws, vws, gws, seq_pad, res_pad, ctx);

  msa_outproj_kernel<<<dim3(NRES * 4), dim3(128), smem3, stream>>>(
      ctx, wTall + (size_t)4 * DMODEL * DMODEL, bo, out);
}
